// SuperPrunedQuanGatePreBasicBlock_64536178589933
// MI455X (gfx1250) — 8-bit matrix operand order repaired and checked by reading, compiled, not run on hardware
//
#include <hip/hip_runtime.h>


typedef __attribute__((ext_vector_type(8))) int v8i;

#define NWELEM  (256 * 256 * 9)
#define KTAP    2304
#define GROUPSZ 4608

#define SZ_ACT   ((size_t)32 * 56 * 56 * 256)
#define SZ_WQ    ((size_t)9 * 256 * 256)
#define OFF_A1   ((size_t)0)
#define OFF_A2   (OFF_A1 + SZ_ACT)
#define OFF_WQ1  (OFF_A2 + SZ_ACT)
#define OFF_WQ2  (OFF_WQ1 + SZ_WQ)
#define OFF_STAT (OFF_WQ2 + SZ_WQ)

__device__ __forceinline__ void async_g2l_b128(const unsigned char* gp, void* lp)
{
    unsigned lds_a = (unsigned)(unsigned long long)lp;
    asm volatile("global_load_async_to_lds_b128 %0, %1, off"
                 :: "v"(lds_a), "v"(gp) : "memory");
}
__device__ __forceinline__ void wait_async_lds()
{
    asm volatile("s_wait_asynccnt 0x0" ::: "memory");
}

__global__ __launch_bounds__(1024)
void wstats_kernel(const float* __restrict__ w1, const float* __restrict__ w2,
                   const float* __restrict__ thr, float* __restrict__ stats)
{
    __shared__ float red[1024];
    __shared__ float gsum[128];
    __shared__ float ind_g[128];
    const int tid = threadIdx.x;

    float lm = 0.f;
    for (int i = tid; i < NWELEM; i += 1024) lm = fmaxf(lm, fabsf(w1[i]));
    red[tid] = lm; __syncthreads();
    for (int s = 512; s > 0; s >>= 1) {
        if (tid < s) red[tid] = fmaxf(red[tid], red[tid + s]);
        __syncthreads();
    }
    const float gmax = red[0]; __syncthreads();

    if (tid < 128) gsum[tid] = 0.f;
    __syncthreads();
    for (int i = tid; i < NWELEM; i += 1024)
        atomicAdd(&gsum[i / GROUPSZ], fabsf(w1[i]));
    __syncthreads();
    const float thv = thr[0];
    if (tid < 128) {
        float gn = (gsum[tid] / gmax) * (1.0f / (float)GROUPSZ);
        ind_g[tid] = (gn > thv) ? 1.f : 0.f;
    }
    __syncthreads();

    red[tid] = (tid < 128) ? ind_g[tid] : 0.f; __syncthreads();
    for (int s = 512; s > 0; s >>= 1) {
        if (tid < s) red[tid] += red[tid + s];
        __syncthreads();
    }
    const float nsel_g = red[0]; __syncthreads();
    const float cnt = nsel_g * 2.f * (float)KTAP;

    float s1 = 0.f, s2 = 0.f;
    for (int i = tid; i < NWELEM; i += 1024) {
        float m = ind_g[(i / KTAP) >> 1];
        float w = w1[i];
        s1 += m * w; s2 += m * w * w;
    }
    red[tid] = s1; __syncthreads();
    for (int s = 512; s > 0; s >>= 1) { if (tid < s) red[tid] += red[tid + s]; __syncthreads(); }
    float S1 = red[0]; __syncthreads();
    red[tid] = s2; __syncthreads();
    for (int s = 512; s > 0; s >>= 1) { if (tid < s) red[tid] += red[tid + s]; __syncthreads(); }
    float S2 = red[0]; __syncthreads();
    float mean1 = S1 / cnt;
    float std1  = sqrtf(fmaxf((S2 - cnt * mean1 * mean1) / (cnt - 1.f), 0.f));

    s1 = 0.f; s2 = 0.f;
    for (int i = tid; i < NWELEM; i += 1024) {
        int cch = (i / 9) & 255;
        float m = ind_g[cch >> 1];
        float w = w2[i];
        s1 += m * w; s2 += m * w * w;
    }
    red[tid] = s1; __syncthreads();
    for (int s = 512; s > 0; s >>= 1) { if (tid < s) red[tid] += red[tid + s]; __syncthreads(); }
    float T1 = red[0]; __syncthreads();
    red[tid] = s2; __syncthreads();
    for (int s = 512; s > 0; s >>= 1) { if (tid < s) red[tid] += red[tid + s]; __syncthreads(); }
    float T2 = red[0]; __syncthreads();
    float mean2 = T1 / cnt;
    float std2  = sqrtf(fmaxf((T2 - cnt * mean2 * mean2) / (cnt - 1.f), 0.f));

    if (tid == 0) {
        stats[0] = mean1; stats[1] = std1;
        stats[2] = mean2; stats[3] = std2;
    }
    if (tid < 256) stats[8 + tid] = ind_g[tid >> 1];
}

__global__ __launch_bounds__(256)
void wquant_kernel(const float* __restrict__ w1, const float* __restrict__ w2,
                   const float* __restrict__ stats,
                   unsigned char* __restrict__ wq1, unsigned char* __restrict__ wq2)
{
    int idx = blockIdx.x * 256 + threadIdx.x;
    if (idx >= 2 * NWELEM) return;
    int sel = idx >= NWELEM;
    int i = idx - sel * NWELEM;
    const float* w = sel ? w2 : w1;
    float mean = stats[sel ? 2 : 0];
    float stdv = stats[sel ? 3 : 1];
    float wn = (w[i] - mean) / (stdv + 1e-6f);
    wn = fminf(fmaxf(wn, -1.f), 1.f);
    unsigned int q = (unsigned int)rintf((wn + 1.f) * 127.5f);
    int o   = i / KTAP;
    int rem = i - o * KTAP;
    int cc  = rem / 9;
    int tp  = rem - cc * 9;
    (sel ? wq2 : wq1)[tp * 65536 + o * 256 + cc] = (unsigned char)q;
}

__global__ __launch_bounds__(256)
void prep1_kernel(const float* __restrict__ x,
                  const float* __restrict__ g, const float* __restrict__ b,
                  const float* __restrict__ m, const float* __restrict__ v,
                  unsigned char* __restrict__ a1)
{
    int idx = blockIdx.x * 256 + threadIdx.x;
    int c = idx & 255;
    int p = idx >> 8;
    int px = p % 56; int t = p / 56;
    int py = t % 56; int bb = t / 56;
    float inv = rsqrtf(v[c] + 1e-5f) * g[c];
    float h = x[((bb * 256 + c) * 56 + py) * 56 + px] * inv + (b[c] - m[c] * inv);
    h = fminf(fmaxf(h, 0.f), 1.f);
    a1[idx] = (unsigned char)(unsigned int)rintf(h * 255.f);
}

template <int PASS>
__global__ __launch_bounds__(256)
void conv_wmma_kernel(const unsigned char* __restrict__ actg,
                      const unsigned char* __restrict__ wq,
                      const float* __restrict__ stats,
                      const float* __restrict__ g2, const float* __restrict__ b2,
                      const float* __restrict__ m2, const float* __restrict__ v2,
                      const float* __restrict__ resid,
                      unsigned char* __restrict__ a2out,
                      float* __restrict__ outp)
{
    extern __shared__ unsigned char smem[];
    unsigned char* actl = smem;
    int* csum = (int*)(smem + 25600);
    int* sa   = (int*)(smem + 26000);
    unsigned char* wbuf0 = smem + 26256;
    unsigned char* wbuf1 = smem + 26256 + 65536;
    unsigned char* wbuf[2] = { wbuf0, wbuf1 };

    const int tid = threadIdx.x;
    const int bb = blockIdx.x / 49;
    const int tt = blockIdx.x - bb * 49;
    const int ty = tt / 7, tx = tt - ty * 7;
    const int y0 = ty * 8, x0 = tx * 8;

    for (int ci = tid; ci < 1600; ci += 256) {
        int pix = ci >> 4;
        int off = (ci & 15) << 4;
        int iy = pix / 10, ix = pix - iy * 10;
        int gy = y0 - 1 + iy, gx = x0 - 1 + ix;
        void* lp = actl + pix * 256 + off;
        if (gy >= 0 && gy < 56 && gx >= 0 && gx < 56) {
            async_g2l_b128(actg + (((bb * 56 + gy) * 56 + gx) << 8) + off, lp);
        } else {
            *(uint4*)lp = make_uint4(0u, 0u, 0u, 0u);
        }
    }
    for (int ci = tid; ci < 4096; ci += 256)
        async_g2l_b128(wq + (ci << 4), wbuf0 + (ci << 4));
    wait_async_lds();
    __syncthreads();

    if (tid < 100) {
        const unsigned int* q = (const unsigned int*)(actl + tid * 256);
        int s = 0;
#pragma unroll 8
        for (int i = 0; i < 64; ++i) {
            unsigned int u = q[i];
            s += (int)((u & 255u) + ((u >> 8) & 255u) + ((u >> 16) & 255u) + (u >> 24));
        }
        csum[tid] = s;
    }
    __syncthreads();
    if (tid < 64) {
        int my = tid >> 3, mx = tid & 7, s = 0;
#pragma unroll
        for (int ky = 0; ky < 3; ++ky)
#pragma unroll
            for (int kx = 0; kx < 3; ++kx)
                s += csum[(my + ky) * 10 + (mx + kx)];
        sa[tid] = s;
    }

    const int wv   = tid >> 5;
    const int lane = tid & 31;
    const int lm   = lane & 15;
    const int half = lane >> 4;

    v8i acc[4][2];
#pragma unroll
    for (int mt = 0; mt < 4; ++mt)
#pragma unroll
        for (int nt = 0; nt < 2; ++nt)
#pragma unroll
            for (int r = 0; r < 8; ++r) acc[mt][nt][r] = 0;

    for (int tap = 0; tap < 9; ++tap) {
        const unsigned char* cw = wbuf[tap & 1];

        if (tap + 1 < 9) {
            unsigned char* nw = wbuf[(tap + 1) & 1];
            for (int ci = tid; ci < 4096; ci += 256)
                async_g2l_b128(wq + (tap + 1) * 65536 + (ci << 4), nw + (ci << 4));
        }
        if (tap + 2 < 9)
            __builtin_prefetch(wq + (tap + 2) * 65536 + tid * 256, 0, 1);

        const int ky = tap / 3, kx = tap - ky * 3;
        const unsigned char* arow[4];
#pragma unroll
        for (int mt = 0; mt < 4; ++mt) {
            int m = mt * 16 + lm;
            int my = m >> 3, mx = m & 7;
            arow[mt] = actl + ((my + ky) * 10 + (mx + kx)) * 256;
        }
        const unsigned char* wrow0 = cw + (wv * 32 + lm) * 256;
        const unsigned char* wrow1 = cw + (wv * 32 + 16 + lm) * 256;

#pragma unroll
        for (int kc = 0; kc < 256; kc += 64) {
            v8i bf0, bf1;
            {
                const uint4* p = (const uint4*)(wrow0 + kc + half * 16);
                uint4 r0 = p[0], r1 = p[2];
                bf0[0] = r0.x; bf0[1] = r0.y; bf0[2] = r0.z; bf0[3] = r0.w;
                bf0[4] = r1.x; bf0[5] = r1.y; bf0[6] = r1.z; bf0[7] = r1.w;
            }
            {
                const uint4* p = (const uint4*)(wrow1 + kc + half * 16);
                uint4 r0 = p[0], r1 = p[2];
                bf1[0] = r0.x; bf1[1] = r0.y; bf1[2] = r0.z; bf1[3] = r0.w;
                bf1[4] = r1.x; bf1[5] = r1.y; bf1[6] = r1.z; bf1[7] = r1.w;
            }
#pragma unroll
            for (int mt = 0; mt < 4; ++mt) {
                const uint4* p = (const uint4*)(arow[mt] + kc + half * 16);
                uint4 r0 = p[0], r1 = p[2];
                v8i af;
                af[0] = r0.x; af[1] = r0.y; af[2] = r0.z; af[3] = r0.w;
                af[4] = r1.x; af[5] = r1.y; af[6] = r1.z; af[7] = r1.w;
                acc[mt][0] = __builtin_amdgcn_wmma_i32_16x16x64_iu8(
                    false, af, false, bf0, acc[mt][0], false, false);
                acc[mt][1] = __builtin_amdgcn_wmma_i32_16x16x64_iu8(
                    false, af, false, bf1, acc[mt][1], false, false);
            }
        }

        wait_async_lds();
        __syncthreads();
    }

    const float inv65025 = 1.0f / 65025.0f;
#pragma unroll
    for (int nt = 0; nt < 2; ++nt) {
        const int n = wv * 32 + nt * 16 + lm;
        float inv = 0.f, bet = 0.f, gate = 1.f;
        if (PASS == 1) {
            inv  = rsqrtf(v2[n] + 1e-5f) * g2[n];
            bet  = b2[n] - m2[n] * inv;
            gate = stats[8 + n];
        }
#pragma unroll
        for (int mt = 0; mt < 4; ++mt) {
#pragma unroll
            for (int r = 0; r < 8; ++r) {
                const int m  = mt * 16 + r + half * 8;
                const int my = m >> 3, mx = m & 7;
                const int gy = y0 + my, gx = x0 + mx;
                const int ival = 2 * acc[mt][nt][r] - 255 * sa[m];
                const float y = (float)ival * inv65025;
                if (PASS == 1) {
                    float h = y * inv + bet;
                    h = fminf(fmaxf(h, 0.f), 1.f);
                    unsigned int qv = (unsigned int)rintf(h * 255.f);
                    if (gate == 0.f) qv = 0u;
                    a2out[(((bb * 56 + gy) * 56 + gx) << 8) + n] = (unsigned char)qv;
                } else {
                    const int oi = ((bb * 256 + n) * 56 + gy) * 56 + gx;
                    outp[oi] = y + resid[oi];
                }
            }
        }
    }
}

extern "C" void kernel_launch(void* const* d_in, const int* in_sizes, int n_in,
                              void* d_out, int out_size, void* d_ws, size_t ws_size,
                              hipStream_t stream)
{
    const float* x   = (const float*)d_in[0];
    const float* g1  = (const float*)d_in[1];
    const float* b1  = (const float*)d_in[2];
    const float* m1  = (const float*)d_in[3];
    const float* v1  = (const float*)d_in[4];
    const float* w1  = (const float*)d_in[5];
    const float* g2  = (const float*)d_in[6];
    const float* b2  = (const float*)d_in[7];
    const float* m2  = (const float*)d_in[8];
    const float* v2  = (const float*)d_in[9];
    const float* w2  = (const float*)d_in[10];
    const float* thr = (const float*)d_in[11];
    (void)in_sizes; (void)n_in; (void)out_size; (void)ws_size;

    unsigned char* ws  = (unsigned char*)d_ws;
    unsigned char* a1  = ws + OFF_A1;
    unsigned char* a2b = ws + OFF_A2;
    unsigned char* wq1 = ws + OFF_WQ1;
    unsigned char* wq2 = ws + OFF_WQ2;
    float* stats       = (float*)(ws + OFF_STAT);

    hipLaunchKernelGGL(wstats_kernel, dim3(1), dim3(1024), 0, stream, w1, w2, thr, stats);
    hipLaunchKernelGGL(wquant_kernel, dim3((2 * NWELEM + 255) / 256), dim3(256), 0, stream,
                       w1, w2, stats, wq1, wq2);
    hipLaunchKernelGGL(prep1_kernel, dim3((int)(SZ_ACT / 256)), dim3(256), 0, stream,
                       x, g1, b1, m1, v1, a1);

    const size_t shmem = 26256 + 2 * 65536;
    hipLaunchKernelGGL((conv_wmma_kernel<1>), dim3(32 * 49), dim3(256), shmem, stream,
                       a1, wq1, stats, g2, b2, m2, v2, (const float*)nullptr,
                       a2b, (float*)nullptr);
    hipLaunchKernelGGL((conv_wmma_kernel<2>), dim3(32 * 49), dim3(256), shmem, stream,
                       a2b, wq2, stats, (const float*)nullptr, (const float*)nullptr,
                       (const float*)nullptr, (const float*)nullptr, x,
                       (unsigned char*)nullptr, (float*)d_out);
}
